// DigitConvolutionalModel_35089882808532
// MI455X (gfx1250) — matrix operand order repaired and checked by reading, not compiled, not run on hardware
//
#include <hip/hip_runtime.h>

// ---------------------------------------------------------------------------
// DigitConvolutionalModel forward on MI455X (gfx1250, wave32, WMMA)
//   x[B,784] -> conv3x3 valid -> [B,676] -> FC1(676->100)+ReLU -> FC2(100->10)
// FC1 GEMM on v_wmma_f32_16x16x32_f16 (f16 in, f32 accumulate).
// Padding: K 676 -> 704 (22 tiles of 32), N 100 -> 112 (7 tiles of 16).
// Workspace (f16):
//   wsA : [B][704]            conv activations, zero-padded K tail
//   wsW : [22][112][32]       fc1_w, K-tile-major so all 7 B-fragments of one
//                             kt fit in the 16-bit DS immediate offset window
//                             (nt stride = 1024 B, kt stride = 7168 B)
// ---------------------------------------------------------------------------

typedef _Float16 v16h __attribute__((ext_vector_type(16)));
typedef _Float16 v8h  __attribute__((ext_vector_type(8)));
typedef float    v8f  __attribute__((ext_vector_type(8)));

#define KPAD 704          // 22 * 32
#define NPAD 112          // 7 * 16
#define KTILES 22
#define NTILES 7

static __device__ __forceinline__ v16h cat16(v8h lo, v8h hi) {
  return __builtin_shufflevector(lo, hi, 0,1,2,3,4,5,6,7,8,9,10,11,12,13,14,15);
}

// ---- Kernel 0: fc1_w (100x676 f32) -> padded f16 [22][112][32] --------------
__global__ __launch_bounds__(256) void wprep_kernel(const float* __restrict__ fc1_w,
                                                    _Float16* __restrict__ wB) {
  int t = blockIdx.x * 256 + threadIdx.x;
  if (t >= KTILES * NPAD * 32) return;
  int kt = t / (NPAD * 32);
  int r  = t - kt * (NPAD * 32);
  int n  = r >> 5;           // 0..111
  int ks = r & 31;           // 0..31
  int k  = kt * 32 + ks;     // 0..703
  float v = (n < 100 && k < 676) ? fc1_w[n * 676 + k] : 0.0f;
  wB[t] = (_Float16)v;
}

// ---- Kernel 1: 3x3 valid conv, f32 compute, f16 padded output ---------------
__global__ __launch_bounds__(256) void conv_kernel(const float* __restrict__ x,
                                                   const float* __restrict__ cw,
                                                   _Float16* __restrict__ A) {
  __shared__ float img[784];
  __shared__ float w[9];
  const int b = blockIdx.x;
  const float* xb = x + (long)b * 784;
  for (int i = threadIdx.x; i < 784; i += 256) img[i] = xb[i];
  if (threadIdx.x < 9) w[threadIdx.x] = cw[threadIdx.x];
  __syncthreads();
  _Float16* Ab = A + (long)b * KPAD;
  for (int k = threadIdx.x; k < KPAD; k += 256) {
    float s = 0.0f;
    if (k < 676) {
      int oy = k / 26;
      int ox = k - oy * 26;
      const float* p = img + oy * 28 + ox;
      s = p[0]  * w[0] + p[1]  * w[1] + p[2]  * w[2]
        + p[28] * w[3] + p[29] * w[4] + p[30] * w[5]
        + p[56] * w[6] + p[57] * w[7] + p[58] * w[8];
    }
    Ab[k] = (_Float16)s;
  }
}

// ---- Kernel 2: FC1 via WMMA + ReLU + FC2, fused ----------------------------
// 256 threads = 8 waves; wave w owns 32 batch rows; block covers 256 rows.
// LDS phase 1 = f16 weights [22][112][32] (157,696 B, CDNA5 320KB LDS);
// LDS phase 2 (same memory) = f32 h [256][112] (114,688 B).
__global__ __launch_bounds__(256) void fc_kernel(const _Float16* __restrict__ convA,
                                                 const _Float16* __restrict__ wB,
                                                 const float* __restrict__ fc1_b,
                                                 const float* __restrict__ fc2_w,
                                                 const float* __restrict__ fc2_b,
                                                 float* __restrict__ out) {
  __shared__ __align__(16) unsigned char smem[KTILES * NPAD * 32 * 2]; // 157,696 B
  _Float16* wlds = (_Float16*)smem;

  const int tid = threadIdx.x;

  // Cooperative stage of the whole padded weight matrix into LDS (b128 chunks).
  {
    const v8h* src = (const v8h*)wB;
    v8h* dst = (v8h*)wlds;
    const int nchunks = (KTILES * NPAD * 32) / 8; // 9856
    for (int c = tid; c < nchunks; c += 256) dst[c] = src[c];
  }
  __syncthreads();

  const int lane = tid & 31;
  const int wave = tid >> 5;
  const int l15  = lane & 15;   // A: M row / B: N col / C: N col
  const int lhi  = lane >> 4;   // K-half selector / C: M+8 selector

  const long rowBase = (long)blockIdx.x * 256 + (long)wave * 32;
  const _Float16* a0p = convA + (rowBase + l15) * KPAD;        // rows 0..15
  const _Float16* a1p = a0p + 16 * (long)KPAD;                 // rows 16..31

  // Per-lane B base inside one kt slab: (l15 rows of 32 halves) + lhi*16 halves.
  // nt adds nt*16*32 = nt*512 halves (1024 B immediate); kt adds 112*32 halves.
  const _Float16* bbase = wlds + l15 * 32 + lhi * 16;

  v8f acc0[NTILES] = {};
  v8f acc1[NTILES] = {};

  for (int kt = 0; kt < KTILES; ++kt) {
    const int ka = kt * 32 + lhi * 16;
    v16h A0 = cat16(*(const v8h*)(a0p + ka), *(const v8h*)(a0p + ka + 8));
    v16h A1 = cat16(*(const v8h*)(a1p + ka), *(const v8h*)(a1p + ka + 8));

    const _Float16* bkt = bbase + kt * (NPAD * 32);
#pragma unroll
    for (int nt = 0; nt < NTILES; ++nt) {
      const _Float16* bp = bkt + nt * (16 * 32);   // +512 halves per nt
      v16h Bf = cat16(*(const v8h*)bp, *(const v8h*)(bp + 8));
      acc0[nt] = __builtin_amdgcn_wmma_f32_16x16x32_f16(
          false, A0, false, Bf, (short)0, acc0[nt], false, false);
      acc1[nt] = __builtin_amdgcn_wmma_f32_16x16x32_f16(
          false, A1, false, Bf, (short)0, acc1[nt], false, false);
    }
  }

  __syncthreads();               // weights dead; reuse LDS for h
  float* hlds = (float*)smem;    // [256][NPAD] f32 = 114,688 B

  // Epilogue 1: bias + ReLU, scatter C fragments to LDS.
  // C layout: VGPR r holds M = r + 8*lhi (per 16-row subtile), N = l15 + 16*nt.
#pragma unroll
  for (int nt = 0; nt < NTILES; ++nt) {
    const int n = nt * 16 + l15;
    const float bias = (n < 100) ? fc1_b[n] : 0.0f;
    const int mBase = wave * 32 + lhi * 8;
#pragma unroll
    for (int r = 0; r < 8; ++r) {
      float v0 = acc0[nt][r] + bias; v0 = v0 > 0.0f ? v0 : 0.0f;
      float v1 = acc1[nt][r] + bias; v1 = v1 > 0.0f ? v1 : 0.0f;
      hlds[(mBase + r) * NPAD + n]      = v0;
      hlds[(mBase + 16 + r) * NPAD + n] = v1;
    }
  }
  __syncthreads();

  // Epilogue 2: FC2 (100 -> 10), one batch row per thread.
  const float* hrow = hlds + tid * NPAD;
  float o[10];
#pragma unroll
  for (int j = 0; j < 10; ++j) o[j] = fc2_b[j];
  for (int n = 0; n < 100; ++n) {
    const float hv = hrow[n];
#pragma unroll
    for (int j = 0; j < 10; ++j) o[j] += hv * fc2_w[j * 100 + n];
  }
  const long m = (long)blockIdx.x * 256 + tid;
  float* op = out + m * 10;
#pragma unroll
  for (int j = 0; j < 10; ++j) op[j] = o[j];
}

// ---------------------------------------------------------------------------
extern "C" void kernel_launch(void* const* d_in, const int* in_sizes, int n_in,
                              void* d_out, int out_size, void* d_ws, size_t ws_size,
                              hipStream_t stream) {
  const float* x      = (const float*)d_in[0];
  const float* conv_w = (const float*)d_in[1];
  const float* fc1_w  = (const float*)d_in[2];
  const float* fc1_b  = (const float*)d_in[3];
  const float* fc2_w  = (const float*)d_in[4];
  const float* fc2_b  = (const float*)d_in[5];
  float* out = (float*)d_out;

  const int B = in_sizes[0] / 784; // 65536

  _Float16* wsA = (_Float16*)d_ws;              // B * 704 halves (~88 MB)
  _Float16* wsW = wsA + (size_t)B * KPAD;       // 22*112*32 halves (157,696 B)

  wprep_kernel<<<(KTILES * NPAD * 32 + 255) / 256, 256, 0, stream>>>(fc1_w, wsW);
  conv_kernel<<<B, 256, 0, stream>>>(x, conv_w, wsA);
  fc_kernel<<<B / 256, 256, 0, stream>>>(wsA, wsW, fc1_b, fc2_w, fc2_b, out);
}
